// Block_84851373900230
// MI455X (gfx1250) — compile-verified, matrix operand order checked by reading, not run on hardware
//
#include <hip/hip_runtime.h>
#include <math.h>

// ---------------------------------------------------------------------------
// Types for CDNA5 WMMA (wave32): V_WMMA_F32_16X16X32_BF16
// ---------------------------------------------------------------------------
typedef __attribute__((ext_vector_type(16))) __bf16         bf16x16;
typedef __attribute__((ext_vector_type(8)))  float          f32x8;
typedef __attribute__((ext_vector_type(8)))  unsigned short u16x8;
typedef __attribute__((ext_vector_type(4)))  unsigned int   u32x4;
typedef __attribute__((ext_vector_type(8)))  int            i32x8;
typedef __attribute__((ext_vector_type(4)))  int            i32x4;

union BfVec {                 // one WMMA 16x32 bf16 operand (8 VGPRs / lane)
    unsigned int u[8];
    bf16x16      bf;
};

__device__ __forceinline__ f32x8 wmma_bf16(bf16x16 a, bf16x16 b, f32x8 c) {
    return __builtin_amdgcn_wmma_f32_16x16x32_bf16(
        false, a, false, b, (short)0, c, false, false);
}

__device__ __forceinline__ f32x8 zero8() {
    f32x8 z;
#pragma unroll
    for (int i = 0; i < 8; ++i) z[i] = 0.f;
    return z;
}

// K index of the first element of dword v for this lane-half (ISA 7.12.2,
// 16-bit A 16x32 layout; B assumed mirrored).
__device__ __forceinline__ int pairK(int v, int half) {
    return ((v & 4) << 2) + (half << 3) + ((v & 3) << 1);
}

__device__ __forceinline__ unsigned short f2bf(float f) {
    union { float f; unsigned int u; } v; v.f = f;
    unsigned int u = v.u;
    return (unsigned short)((u + 0x7FFFu + ((u >> 16) & 1u)) >> 16);   // RNE
}

// exp-based tanh GELU: x * e/(e+1), e = exp(2*0.79788456*(x+0.044715 x^3))
__device__ __forceinline__ float gelu_tanh(float x) {
    float u = 0.7978845608028654f * (x + 0.044715f * x * x * x);
    float e = __expf(fminf(2.0f * u, 80.0f));
    return x * e / (e + 1.0f);
}

// ---------------------------------------------------------------------------
// TDM: 2D tensor tile -> LDS (D# per ISA 8.3/8.4).  data_size=2B.
// ---------------------------------------------------------------------------
#if __has_builtin(__builtin_amdgcn_tensor_load_to_lds)
#define USE_TDM 1
#else
#define USE_TDM 0
#endif

#if USE_TDM
__device__ __forceinline__ void tdm_load_2d_bf16(
    unsigned int lds_addr, const void* gptr,
    unsigned int tensor_d0, unsigned int tensor_d1,
    unsigned int tile_d0, unsigned int tile_d1,
    unsigned long long stride0_elems) {
    unsigned long long ga = (unsigned long long)gptr;
    u32x4 g0;
    g0[0] = 1u;                                   // count=1, user descriptor
    g0[1] = lds_addr;                             // LDS byte address
    g0[2] = (unsigned int)ga;                     // global_addr[31:0]
    g0[3] = (unsigned int)(ga >> 32) | (2u << 30);// global_addr[56:32] | type=2
    i32x8 g1;
    g1[0] = (int)(1u << 16);                      // data_size=1 (2 bytes)
    g1[1] = (int)((tensor_d0 & 0xFFFFu) << 16);   // tensor_dim0[15:0]
    g1[2] = (int)((tensor_d0 >> 16) | ((tensor_d1 & 0xFFFFu) << 16));
    g1[3] = (int)((tensor_d1 >> 16) | (tile_d0 << 16));
    g1[4] = (int)(tile_d1 & 0xFFFFu);             // tile_dim1 (tile_dim2=0)
    g1[5] = (int)(unsigned int)(stride0_elems & 0xFFFFFFFFull);
    g1[6] = (int)(unsigned int)(stride0_elems >> 32);
    g1[7] = 0;
    i32x4 z = {0, 0, 0, 0};
#if defined(__clang_major__) && (__clang_major__ >= 23)
    i32x8 z8 = {0, 0, 0, 0, 0, 0, 0, 0};
    __builtin_amdgcn_tensor_load_to_lds(g0, g1, z, z, z8, 0);
#else
    __builtin_amdgcn_tensor_load_to_lds(g0, g1, z, z, 0);
#endif
}
__device__ __forceinline__ void tdm_wait0() {
#if __has_builtin(__builtin_amdgcn_s_wait_tensorcnt)
    __builtin_amdgcn_s_wait_tensorcnt(0);
#else
    asm volatile("s_wait_tensorcnt 0x0" ::: "memory");
#endif
}
#endif

// ---------------------------------------------------------------------------
// fp32 -> bf16 conversion (weights)
// ---------------------------------------------------------------------------
__global__ __launch_bounds__(256) void cvt_f32_bf16_kernel(
    const float* __restrict__ in, unsigned short* __restrict__ out, int n) {
    int i = blockIdx.x * 256 + threadIdx.x;
    if (i < n) out[i] = f2bf(in[i]);
}

// ---------------------------------------------------------------------------
// LayerNorm over C, bf16 output.  One block (256 thr) per row.
// ---------------------------------------------------------------------------
__global__ __launch_bounds__(256) void ln_bf16_kernel(
    const float* __restrict__ x, const float* __restrict__ w,
    const float* __restrict__ b, unsigned short* __restrict__ out, int C) {
    int row = blockIdx.x, tid = threadIdx.x;
    const float* xr = x + (size_t)row * C;
    __shared__ float red[256];

    float s = 0.f;
    for (int c = tid; c < C; c += 256) s += xr[c];
    red[tid] = s; __syncthreads();
#pragma unroll
    for (int st = 128; st > 0; st >>= 1) {
        if (tid < st) red[tid] += red[tid + st];
        __syncthreads();
    }
    float mu = red[0] / (float)C;
    __syncthreads();

    float sq = 0.f;
    for (int c = tid; c < C; c += 256) { float d = xr[c] - mu; sq += d * d; }
    red[tid] = sq; __syncthreads();
#pragma unroll
    for (int st = 128; st > 0; st >>= 1) {
        if (tid < st) red[tid] += red[tid + st];
        __syncthreads();
    }
    float rstd = rsqrtf(red[0] / (float)C + 1e-5f);

    unsigned short* orow = out + (size_t)row * C;
    for (int c = tid; c < C; c += 256)
        orow[c] = f2bf((xr[c] - mu) * rstd * w[c] + b[c]);
}

// ---------------------------------------------------------------------------
// Tiled WMMA GEMM:  out = act(A[M,K](bf16) @ B[K,N](bf16) + bias) + res
// Block tile 64x256, K-chunk 32, 8 waves (2M x 4N), wave tile 32x64.
// Double-buffered LDS software pipeline:
//   - A chunk i+1 staged by the Tensor Data Mover, overlapped with compute i
//   - B chunk i+1 prefetched to registers before compute i, stored after.
// ---------------------------------------------------------------------------
__global__ __launch_bounds__(256) void gemm_bf16_kernel(
    const unsigned short* __restrict__ A, const unsigned short* __restrict__ Bm,
    const float* __restrict__ bias, const float* __restrict__ res,
    float* __restrict__ outF, unsigned short* __restrict__ outH,
    int M, int N, int K, int act) {
    __shared__ __align__(16) unsigned short As[2][64 * 32];   // [m][k]
    __shared__ __align__(16) unsigned short Bs[2][256 * 32];  // [n][k] transp.

    const int tid  = threadIdx.x;
    const int lane = tid & 31, w = tid >> 5;
    const int half = lane >> 4, lm = lane & 15;
    const int wm = w >> 2, wn = w & 3;                 // 2 x 4 wave grid
    const int m0 = blockIdx.y * 64, n0 = blockIdx.x * 256;

    f32x8 acc[2][4];
#pragma unroll
    for (int mt = 0; mt < 2; ++mt)
#pragma unroll
        for (int nt = 0; nt < 4; ++nt) acc[mt][nt] = zero8();

    const int bkk = tid >> 3, bnb = (tid & 7) << 5;    // B: 32 elems/thread
#if !USE_TDM
    const int arow = tid >> 2, acb = (tid & 3) << 3;   // A: 8 elems/thread
#endif

    // ---------------- prologue: stage chunk 0 into buffer 0 ----------------
#if USE_TDM
    if (tid == 0)
        tdm_load_2d_bf16((unsigned int)(size_t)(void*)&As[0][0],
                         A + (size_t)m0 * K,
                         (unsigned)K, (unsigned)(M - m0), 32u, 64u,
                         (unsigned long long)K);
#else
    *(u16x8*)&As[0][arow * 32 + acb] =
        *(const u16x8*)(A + (size_t)(m0 + arow) * K + acb);
#endif
    {
        const unsigned short* bsrc = Bm + (size_t)bkk * N + n0 + bnb;
        u16x8 pb[4];
#pragma unroll
        for (int q = 0; q < 4; ++q) pb[q] = *(const u16x8*)(bsrc + q * 8);
#pragma unroll
        for (int q = 0; q < 4; ++q)
#pragma unroll
            for (int j = 0; j < 8; ++j)
                Bs[0][(bnb + q * 8 + j) * 32 + bkk] = pb[q][j];
    }
#if USE_TDM
    if (tid == 0) tdm_wait0();
#endif
    __syncthreads();

    const int nIter = K >> 5;
    for (int it = 0; it < nIter; ++it) {
        const int cur = it & 1, nxt = cur ^ 1;
        const int k0 = it << 5;
        const bool hasNext = (it + 1 < nIter);

        // ---- kick off next chunk: TDM for A, register-prefetch for B ----
        u16x8 pb[4];
        if (hasNext) {
#if USE_TDM
            if (tid == 0)
                tdm_load_2d_bf16((unsigned int)(size_t)(void*)&As[nxt][0],
                                 A + (size_t)m0 * K + k0 + 32,
                                 (unsigned)(K - k0 - 32), (unsigned)(M - m0),
                                 32u, 64u, (unsigned long long)K);
#else
            pbA_load:;
#endif
            const unsigned short* bsrc =
                Bm + (size_t)(k0 + 32 + bkk) * N + n0 + bnb;
#pragma unroll
            for (int q = 0; q < 4; ++q) pb[q] = *(const u16x8*)(bsrc + q * 8);
        }
#if !USE_TDM
        u16x8 pa;
        if (hasNext)
            pa = *(const u16x8*)(A + (size_t)(m0 + arow) * K + k0 + 32 + acb);
#endif

        // ---- compute on current buffer ----
        const unsigned short* Ac = &As[cur][0];
        const unsigned short* Bc = &Bs[cur][0];
        BfVec a[2], bvv[4];
#pragma unroll
        for (int mt = 0; mt < 2; ++mt) {
            int row = wm * 32 + mt * 16 + lm;
#pragma unroll
            for (int v = 0; v < 8; ++v)
                a[mt].u[v] = *(const unsigned int*)&Ac[row * 32 + pairK(v, half)];
        }
#pragma unroll
        for (int nt = 0; nt < 4; ++nt) {
            int nn = wn * 64 + nt * 16 + lm;
#pragma unroll
            for (int v = 0; v < 8; ++v)
                bvv[nt].u[v] = *(const unsigned int*)&Bc[nn * 32 + pairK(v, half)];
        }
#pragma unroll
        for (int mt = 0; mt < 2; ++mt)
#pragma unroll
            for (int nt = 0; nt < 4; ++nt)
                acc[mt][nt] = wmma_bf16(a[mt].bf, bvv[nt].bf, acc[mt][nt]);

        // ---- drain next-chunk staging into the other buffer ----
        if (hasNext) {
#pragma unroll
            for (int q = 0; q < 4; ++q)
#pragma unroll
                for (int j = 0; j < 8; ++j)
                    Bs[nxt][(bnb + q * 8 + j) * 32 + bkk] = pb[q][j];
#if !USE_TDM
            *(u16x8*)&As[nxt][arow * 32 + acb] = pa;
#else
            if (tid == 0) tdm_wait0();
#endif
        }
        __syncthreads();
    }

    // ---- epilogue: bias, activation, residual, stores ----
#pragma unroll
    for (int mt = 0; mt < 2; ++mt)
#pragma unroll
        for (int nt = 0; nt < 4; ++nt)
#pragma unroll
            for (int r = 0; r < 8; ++r) {
                int m = m0 + wm * 32 + mt * 16 + r + 8 * half;
                int n = n0 + wn * 64 + nt * 16 + lm;
                float v = acc[mt][nt][r];
                if (bias) v += bias[n];
                if (act == 1) v = gelu_tanh(v);
                if (res)  v += res[(size_t)m * N + n];
                if (outF) outF[(size_t)m * N + n] = v;
                if (outH) outH[(size_t)m * N + n] = f2bf(v);
            }
}

// ---------------------------------------------------------------------------
// Flash attention (causal), one wave per (b, h, 16-row Q tile).
// Q tile 16x64, KV chunks of 32, online softmax, P@V via WMMA.
// V chunk staged row-major with async global->LDS b128 (ASYNCcnt path),
// overlapped with the Q@K^T WMMAs and the softmax update.
// ---------------------------------------------------------------------------
__global__ __launch_bounds__(32) void flash_attn_kernel(
    const unsigned short* __restrict__ qkvh, unsigned short* __restrict__ yh,
    int T, int C, int H) {
    const int D  = 64;
    const int b  = blockIdx.z, hh = blockIdx.y, q0 = blockIdx.x * 16;
    const int lane = threadIdx.x;
    const int half = lane >> 4, lm = lane & 15;
    const int ld = 3 * C;

    const unsigned short* base = qkvh + (size_t)b * T * ld;
    const unsigned short* qp = base +         hh * D;
    const unsigned short* kp = base + C +     hh * D;
    const unsigned short* vp = base + 2 * C + hh * D;

    __shared__ __align__(16) unsigned short Ps[16 * 32];   // P tile  [m][kk]
    __shared__ __align__(16) unsigned short Vs[32 * 64];   // V chunk [kk][d]
    const unsigned int vsBase = (unsigned int)(size_t)(void*)Vs;

    // ---- load Q A-operands (16x64 = two 16x32 regs), once ----
    BfVec QA[2];
#pragma unroll
    for (int ri = 0; ri < 2; ++ri) {
        const unsigned short* qrow = qp + (size_t)(q0 + lm) * ld + ri * 32;
#pragma unroll
        for (int v = 0; v < 8; ++v)
            QA[ri].u[v] = *(const unsigned int*)(qrow + pairK(v, half));
    }

    f32x8 O[4];
#pragma unroll
    for (int dt = 0; dt < 4; ++dt) O[dt] = zero8();
    float mi[8], li[8];
#pragma unroll
    for (int r = 0; r < 8; ++r) { mi[r] = -1e30f; li[r] = 0.f; }

    for (int kv0 = 0; kv0 <= q0 + 15; kv0 += 32) {
        // previous chunk's DS reads must retire before async rewrites Vs
        asm volatile("s_wait_dscnt 0x0" ::: "memory");

        // ---- async-stage V chunk: Vs[kk][d], 16B per lane per op ----
        {
            const unsigned short* vrow = vp + (size_t)(kv0 + lane) * ld;
            unsigned int ldsRow = vsBase + (unsigned int)lane * 128u;
#pragma unroll
            for (int d8 = 0; d8 < 64; d8 += 8) {
                unsigned long long ga = (unsigned long long)(vrow + d8);
                unsigned int la = ldsRow + (unsigned int)d8 * 2u;
                asm volatile("global_load_async_to_lds_b128 %0, %1, off"
                             :: "v"(la), "v"(ga) : "memory");
            }
        }

        // ---- scores S = Q @ K^T (K^T operand is K-contiguous in memory) ----
        f32x8 S[2];
#pragma unroll
        for (int nt = 0; nt < 2; ++nt) {
            BfVec kb0, kb1;
            const unsigned short* krow = kp + (size_t)(kv0 + nt * 16 + lm) * ld;
#pragma unroll
            for (int v = 0; v < 8; ++v) {
                kb0.u[v] = *(const unsigned int*)(krow +      pairK(v, half));
                kb1.u[v] = *(const unsigned int*)(krow + 32 + pairK(v, half));
            }
            f32x8 a = zero8();
            a = wmma_bf16(QA[0].bf, kb0.bf, a);
            a = wmma_bf16(QA[1].bf, kb1.bf, a);
            S[nt] = a;
        }

        // ---- scale, causal mask, online softmax update ----
        float mnew[8], alpha[8];
#pragma unroll
        for (int r = 0; r < 8; ++r) {
            int row = q0 + r + 8 * half;
            float s0 = S[0][r] * 0.125f;
            float s1 = S[1][r] * 0.125f;
            if (kv0 + lm      > row) s0 = -1e30f;
            if (kv0 + 16 + lm > row) s1 = -1e30f;
            S[0][r] = s0; S[1][r] = s1;
            float mx = fmaxf(s0, s1);
#pragma unroll
            for (int d = 1; d < 16; d <<= 1) mx = fmaxf(mx, __shfl_xor(mx, d, 32));
            mnew[r]  = fmaxf(mi[r], mx);
            alpha[r] = __expf(mi[r] - mnew[r]);
            mi[r]    = mnew[r];
        }
#pragma unroll
        for (int r = 0; r < 8; ++r) {
            float p0 = __expf(S[0][r] - mnew[r]);
            float p1 = __expf(S[1][r] - mnew[r]);
            int m = r + 8 * half;
            Ps[m * 32 +      lm] = f2bf(p0);
            Ps[m * 32 + 16 + lm] = f2bf(p1);
            float rs = p0 + p1;
#pragma unroll
            for (int d = 1; d < 16; d <<= 1) rs += __shfl_xor(rs, d, 32);
            li[r] = li[r] * alpha[r] + rs;
#pragma unroll
            for (int dt = 0; dt < 4; ++dt) O[dt][r] *= alpha[r];
        }
        __syncthreads();

        // ---- O += P(16x32) @ V(32x64) ----
        BfVec PA;
#pragma unroll
        for (int v = 0; v < 8; ++v)
            PA.u[v] = *(const unsigned int*)&Ps[lm * 32 + pairK(v, half)];

        asm volatile("s_wait_asynccnt 0x0" ::: "memory");   // V in LDS now
#pragma unroll
        for (int dt = 0; dt < 4; ++dt) {
            BfVec vb;
#pragma unroll
            for (int v = 0; v < 8; ++v) {
                int kk = pairK(v, half);
                int n  = dt * 16 + lm;
                unsigned int lo = Vs[kk * 64 + n];
                unsigned int hi = Vs[(kk + 1) * 64 + n];
                vb.u[v] = lo | (hi << 16);
            }
            O[dt] = wmma_bf16(PA.bf, vb.bf, O[dt]);
        }
    }

    // ---- normalize and store y (bf16, [B*T, C]) ----
#pragma unroll
    for (int r = 0; r < 8; ++r) {
        float inv = 1.0f / li[r];
        int row = q0 + r + 8 * half;
#pragma unroll
        for (int dt = 0; dt < 4; ++dt) {
            int col = hh * D + dt * 16 + lm;
            yh[(size_t)(b * T + row) * C + col] = f2bf(O[dt][r] * inv);
        }
    }
}

// ---------------------------------------------------------------------------
// Host orchestration
// ---------------------------------------------------------------------------
extern "C" void kernel_launch(void* const* d_in, const int* in_sizes, int n_in,
                              void* d_out, int out_size, void* d_ws, size_t ws_size,
                              hipStream_t stream) {
    (void)in_sizes; (void)n_in; (void)out_size; (void)ws_size;
    const float* x     = (const float*)d_in[0];
    const float* ln1_w = (const float*)d_in[1];
    const float* ln1_b = (const float*)d_in[2];
    const float* qkv_w = (const float*)d_in[3];
    const float* qkv_b = (const float*)d_in[4];
    const float* out_w = (const float*)d_in[5];
    const float* out_b = (const float*)d_in[6];
    const float* fc1_w = (const float*)d_in[7];
    const float* fc1_b = (const float*)d_in[8];
    const float* fc2_w = (const float*)d_in[9];
    const float* fc2_b = (const float*)d_in[10];
    float* out = (float*)d_out;

    const int B = 2, T = 2048, C = 768, H = 12;
    const int M = B * T;                       // 4096

    char* ws = (char*)d_ws;
    size_t off = 0;
    auto alloc = [&](size_t bytes) -> char* {
        char* p = ws + off;
        off += (bytes + 255) & ~(size_t)255;
        return p;
    };
    unsigned short* hb    = (unsigned short*)alloc((size_t)M * C * 2);
    unsigned short* qkvWh = (unsigned short*)alloc((size_t)C * 3 * C * 2);
    unsigned short* outWh = (unsigned short*)alloc((size_t)C * C * 2);
    unsigned short* fc1Wh = (unsigned short*)alloc((size_t)C * 4 * C * 2);
    unsigned short* fc2Wh = (unsigned short*)alloc((size_t)4 * C * C * 2);
    unsigned short* qkvh  = (unsigned short*)alloc((size_t)M * 3 * C * 2);
    unsigned short* yh    = (unsigned short*)alloc((size_t)M * C * 2);
    float*          x1    = (float*)        alloc((size_t)M * C * 4);
    unsigned short* h2b   = (unsigned short*)alloc((size_t)M * C * 2);
    unsigned short* gh    = (unsigned short*)alloc((size_t)M * 4 * C * 2);

    auto cvt = [&](const float* src, unsigned short* dst, int n) {
        cvt_f32_bf16_kernel<<<(n + 255) / 256, 256, 0, stream>>>(src, dst, n);
    };
    cvt(qkv_w, qkvWh, C * 3 * C);
    cvt(out_w, outWh, C * C);
    cvt(fc1_w, fc1Wh, C * 4 * C);
    cvt(fc2_w, fc2Wh, 4 * C * C);

    // h = LN(x)
    ln_bf16_kernel<<<M, 256, 0, stream>>>(x, ln1_w, ln1_b, hb, C);

    // qkv = h @ qkv_w + qkv_b   (bf16 out for attention)
    gemm_bf16_kernel<<<dim3(3 * C / 256, M / 64), 256, 0, stream>>>(
        hb, qkvWh, qkv_b, nullptr, nullptr, qkvh, M, 3 * C, C, 0);

    // y = causal_attention(q, k, v)   (bf16)
    flash_attn_kernel<<<dim3(T / 16, H, B), 32, 0, stream>>>(qkvh, yh, T, C, H);

    // x1 = x + y @ out_w + out_b   (fp32)
    gemm_bf16_kernel<<<dim3(C / 256, M / 64), 256, 0, stream>>>(
        yh, outWh, out_b, x, x1, nullptr, M, C, C, 0);

    // h2 = LN(x1)  (reference re-applies ln_1)
    ln_bf16_kernel<<<M, 256, 0, stream>>>(x1, ln1_w, ln1_b, h2b, C);

    // g = gelu(h2 @ fc1_w + fc1_b)   (bf16)
    gemm_bf16_kernel<<<dim3(4 * C / 256, M / 64), 256, 0, stream>>>(
        h2b, fc1Wh, fc1_b, nullptr, nullptr, gh, M, 4 * C, C, 1);

    // out = x1 + g @ fc2_w + fc2_b   (fp32)
    gemm_bf16_kernel<<<dim3(C / 256, M / 64), 256, 0, stream>>>(
        gh, fc2Wh, fc2_b, x1, out, nullptr, M, C, 4 * C, 0);
}
